// IG_MSA_26087631356177
// MI455X (gfx1250) — matrix operand order repaired and checked by reading, not compiled, not run on hardware
//
#include <hip/hip_runtime.h>
#include <math.h>

typedef _Float16 f16;
typedef __attribute__((ext_vector_type(16))) _Float16 v16h;
typedef __attribute__((ext_vector_type(8)))  _Float16 v8h;
typedef __attribute__((ext_vector_type(4)))  _Float16 v4h;
typedef __attribute__((ext_vector_type(8)))  float    v8f;
typedef __attribute__((ext_vector_type(4)))  float    v4f;

#define NTOK 65536      // H*W
#define XS_STRIDE 136   // 128 tok + 8 pad (f16): 68-dword row stride, conflict-free
#define WV_STRIDE 72    // 64 + 8 pad
#define WP_STRIDE 264   // 256 + 8 pad

static __device__ __forceinline__ v8f vzero8() {
  v8f r;
#pragma unroll
  for (int i = 0; i < 8; ++i) r[i] = 0.0f;
  return r;
}

static __device__ __forceinline__ v16h combine16(v8h a, v8h b) {
  v16h r;
#pragma unroll
  for (int j = 0; j < 8; ++j) { r[j] = a[j]; r[8 + j] = b[j]; }
  return r;
}

// 16 consecutive f16 from LDS as two 16B ds_load_b128
static __device__ __forceinline__ v16h ld16(const f16* p) {
  return combine16(*(const v8h*)p, *(const v8h*)(p + 8));
}

static __device__ __forceinline__ v8f wmma16(v16h a, v16h b, v8f c) {
  // emits v_wmma_f32_16x16x32_f16
  return __builtin_amdgcn_wmma_f32_16x16x32_f16(
      /*neg_a=*/false, a, /*neg_b=*/false, b,
      /*c_mod=*/(short)0, c, /*reuse_a=*/false, /*reuse_b=*/false);
}

// ---------------------------------------------------------------------------
// Pass 1: per-batch Gram matrix S = X^T X (64x64), contraction over tokens.
// Block = 256 threads (8 waves) stages a 64ch x 128tok chunk of x into LDS
// (f16, padded rows, coalesced v4f reads). The 4x4 C-tile grid is split over
// 8 waves (2 tiles per wave -> 16 accum VGPRs, 2 waves/SIMD occupancy).
// Fragments are batched into registers before the WMMA chain so DS loads
// clause up and WMMAs issue back-to-back.
// Grid: dim3(32, 4) -> 2048 tokens/block, 16 chunks.
// ---------------------------------------------------------------------------
__global__ void ig_pass1_gram(const float* __restrict__ x,
                              float* __restrict__ partials)
{
  __shared__ f16 xs[64 * XS_STRIDE];   // ~17 KB

  const int lane = threadIdx.x & 31;
  const int wv   = threadIdx.x >> 5;   // 0..7
  const int b    = blockIdx.y;
  const int m    = lane & 15;
  const int hi   = lane >> 4;
  const int mi   = wv >> 1;            // M-strip (0..3)
  const int nh   = (wv & 1) * 2;       // first of two N-strips
  const float* xb = x + ((size_t)b << 22);          // b*64*65536
  const int tok0 = blockIdx.x * 2048;

  v8f acc[2];
#pragma unroll
  for (int j = 0; j < 2; ++j) acc[j] = vzero8();

  for (int chunk = 0; chunk < 16; ++chunk) {
    const int tokc = tok0 + chunk * 128;
    // cooperative stage: 64ch x 128tok f32 -> f16 LDS (2048 v4f / 256 thr)
    for (int i = threadIdx.x; i < 64 * 32; i += blockDim.x) {
      int ch = i >> 5;
      int cv = (i & 31) * 4;
      v4f t = *(const v4f*)(xb + ((size_t)ch << 16) + tokc + cv);
      v4h hm;
#pragma unroll
      for (int j = 0; j < 4; ++j) hm[j] = (f16)t[j];
      *(v4h*)(xs + ch * XS_STRIDE + cv) = hm;
    }
    __syncthreads();

#pragma unroll
    for (int ks = 0; ks < 4; ++ks) {
      const int kt = ks * 32;
      const f16* ar = xs + (mi * 16 + m) * XS_STRIDE + kt;
      v16h A = ld16(ar + hi * 16);
      v16h Bt[2];
#pragma unroll
      for (int j = 0; j < 2; ++j)
        Bt[j] = ld16(xs + ((nh + j) * 16 + m) * XS_STRIDE + kt + hi * 16);
#pragma unroll
      for (int j = 0; j < 2; ++j)
        acc[j] = wmma16(A, Bt[j], acc[j]);
    }
    __syncthreads();
  }

  // write partial S in semantic [64][64] layout
  float* P = partials + ((size_t)(b * 32 + blockIdx.x) << 12);
#pragma unroll
  for (int j = 0; j < 2; ++j)
#pragma unroll
    for (int r = 0; r < 8; ++r)
      P[(mi * 16 + r + 8 * hi) * 64 + (nh + j) * 16 + m] = acc[j][r];
}

// Deterministic fixed-order reduction of the 32 block-partials per batch.
__global__ void ig_pass1_reduce(const float* __restrict__ partials,
                                float* __restrict__ S)
{
  int idx = blockIdx.x * blockDim.x + threadIdx.x;   // 0..16383
  int b = idx >> 12;
  int e = idx & 4095;
  float s = 0.0f;
  for (int p = 0; p < 32; ++p)
    s += partials[(((size_t)(b * 32 + p)) << 12) + e];
  S[idx] = s;
}

// ---------------------------------------------------------------------------
// Pass 1.5: per-batch tiny kernel (4 blocks x 64 threads).
//   G_h = Wk_h^T S Wq_h ; norms from diag(W^T S W)
//   attn = softmax_rows(G / (||k_d|| ||q_e||) * rescale_h)
//   Wp2T[c][h*64+e] = sum_d attn[d][e] * Wp[h*64+d][c]  (stored f16)
// ---------------------------------------------------------------------------
__global__ void ig_pass_attn(const float* __restrict__ Sg,
                             const float* __restrict__ Wq,
                             const float* __restrict__ Wk,
                             const float* __restrict__ Wp,
                             const float* __restrict__ rescale,
                             f16* __restrict__ Wp2T)
{
  __shared__ float S[64][64];
  __shared__ float G[64][64];
  __shared__ float Dq[64];
  __shared__ float Dk[64];
  const int t = threadIdx.x;     // 0..63
  const int b = blockIdx.x;

  for (int i = t; i < 4096; i += 64) (&S[0][0])[i] = Sg[(b << 12) + i];
  __syncthreads();

  for (int h = 0; h < 4; ++h) {
    float wq[64], wk[64];
    for (int c = 0; c < 64; ++c) {
      wq[c] = Wq[c * 256 + h * 64 + t];
      wk[c] = Wk[c * 256 + h * 64 + t];
    }
    float Tq[64], Tk[64];
    for (int c = 0; c < 64; ++c) {
      float aq = 0.0f, ak = 0.0f;
      for (int cp = 0; cp < 64; ++cp) {
        float s = S[c][cp];
        aq += s * wq[cp];
        ak += s * wk[cp];
      }
      Tq[c] = aq; Tk[c] = ak;
    }
    float dq = 0.0f, dk = 0.0f;
    for (int c = 0; c < 64; ++c) { dq += wq[c] * Tq[c]; dk += wk[c] * Tk[c]; }
    Dq[t] = dq; Dk[t] = dk;
    for (int d = 0; d < 64; ++d) {
      float g = 0.0f;
      for (int c = 0; c < 64; ++c) g += Wk[c * 256 + h * 64 + d] * Tq[c];
      G[d][t] = g;
    }
    __syncthreads();

    {
      float nk = fmaxf(sqrtf(fmaxf(Dk[t], 0.0f)), 1e-12f);
      float rs = rescale[h];
      float rowv[64];
      float mx = -3.0e38f;
      for (int e = 0; e < 64; ++e) {
        float nq = fmaxf(sqrtf(fmaxf(Dq[e], 0.0f)), 1e-12f);
        float a = G[t][e] / (nk * nq) * rs;
        rowv[e] = a;
        mx = fmaxf(mx, a);
      }
      float ssum = 0.0f;
      for (int e = 0; e < 64; ++e) { float p = expf(rowv[e] - mx); rowv[e] = p; ssum += p; }
      float inv = 1.0f / ssum;
      for (int e = 0; e < 64; ++e) G[t][e] = rowv[e] * inv;   // row t only: no race
    }
    __syncthreads();

    {
      float wp[64];
      for (int d = 0; d < 64; ++d) wp[d] = Wp[(h * 64 + d) * 64 + t];
      for (int e = 0; e < 64; ++e) {
        float a = 0.0f;
        for (int d = 0; d < 64; ++d) a += G[d][e] * wp[d];
        Wp2T[(((size_t)b * 64 + t) << 8) + h * 64 + e] = (f16)a;
      }
    }
    __syncthreads();
  }
}

// ---------------------------------------------------------------------------
// Pass 2: streaming fused v-projection + attention apply + output projection.
// Per 16-token tile (per wave): vfull = (X Wv) .* (Illu WL)  [16x256] via
// 64 WMMAs, elementwise product in C-layout, f16 round-trip through a
// per-wave padded LDS buffer (C->A re-layout), then 32 WMMAs for
// y = vfull @ Wp2 + bp with coalesced b128 stores to NCHW output.
// ---------------------------------------------------------------------------
__global__ void ig_pass2(const float* __restrict__ x,
                         const float* __restrict__ illu,
                         const float* __restrict__ Wv,
                         const float* __restrict__ WL,
                         const f16* __restrict__ Wp2T,
                         const float* __restrict__ bp,
                         float* __restrict__ out)
{
  extern __shared__ char smem[];
  f16*   wvT  = (f16*)smem;                         // [256][WV_STRIDE]
  f16*   wlT  = wvT + 256 * WV_STRIDE;
  f16*   wp2  = wlT + 256 * WV_STRIDE;              // [64][WP_STRIDE]
  float* bps  = (float*)(wp2 + 64 * WP_STRIDE);
  f16*   vbase = (f16*)(bps + 64);                  // 4 waves x [16][WP_STRIDE]

  const int lane = threadIdx.x & 31;
  const int wv_  = threadIdx.x >> 5;
  const int b    = blockIdx.y;
  const int m    = lane & 15;
  const int hi   = lane >> 4;

  // stage weights (transpose Wv/WL so K is contiguous per B-row)
  for (int i = threadIdx.x; i < 64 * 256; i += blockDim.x) {
    int k = i >> 8, n = i & 255;            // i = k*256 + n (coalesced reads)
    wvT[n * WV_STRIDE + k] = (f16)Wv[i];
    wlT[n * WV_STRIDE + k] = (f16)WL[i];
    wp2[k * WP_STRIDE + n] = Wp2T[((size_t)b << 14) + i];   // k=row(c), n=col(j)
  }
  for (int i = threadIdx.x; i < 64; i += blockDim.x) bps[i] = bp[i];
  __syncthreads();

  f16* vbuf = vbase + wv_ * (16 * WP_STRIDE);
  const float* xb = x    + ((size_t)b << 22);
  const float* ib = illu + ((size_t)b << 22);
  float*       ob = out  + ((size_t)b << 22);

  const int wavesPerBatch = gridDim.x * (blockDim.x >> 5);
  for (int tile = blockIdx.x * (blockDim.x >> 5) + wv_; tile < 4096;
       tile += wavesPerBatch) {
    const int tokb = tile << 4;

    // A fragments for x and illu: A(m=token, k=channel), two K-blocks of 32
    v16h Ax[2], Ai[2];
#pragma unroll
    for (int kb = 0; kb < 2; ++kb) {
      v16h A, I;
#pragma unroll
      for (int j = 0; j < 16; ++j) {
        int k = kb * 32 + hi * 16 + j;   // channel
        size_t off = ((size_t)k << 16) + tokb + m;
        A[j] = (f16)xb[off];
        I[j] = (f16)ib[off];
      }
      Ax[kb] = A; Ai[kb] = I;
    }

    // GEMM 1+2 fused: vfull tiles, elementwise product in C-layout, f16->LDS
#pragma unroll 2
    for (int nb = 0; nb < 16; ++nb) {
      const f16* wvr = wvT + (nb * 16 + m) * WV_STRIDE;
      const f16* wlr = wlT + (nb * 16 + m) * WV_STRIDE;
      v16h Bv0 = ld16(wvr + hi * 16);
      v16h Bv1 = ld16(wvr + 32 + hi * 16);
      v16h Bl0 = ld16(wlr + hi * 16);
      v16h Bl1 = ld16(wlr + 32 + hi * 16);
      v8f p = vzero8();
      v8f q = vzero8();
      p = wmma16(Ax[0], Bv0, p);
      p = wmma16(Ax[1], Bv1, p);
      q = wmma16(Ai[0], Bl0, q);
      q = wmma16(Ai[1], Bl1, q);
#pragma unroll
      for (int r = 0; r < 8; ++r)
        vbuf[(r + 8 * hi) * WP_STRIDE + nb * 16 + m] = (f16)(p[r] * q[r]);
    }

    // GEMM 3: y[16 tok x 64 ch] = vfull[16x256] @ Wp2[256x64]
    v8f y[4];
#pragma unroll
    for (int nb = 0; nb < 4; ++nb) y[nb] = vzero8();
#pragma unroll 2
    for (int kb = 0; kb < 8; ++kb) {
      const f16* vr = vbuf + m * WP_STRIDE + kb * 32;
      v16h A = ld16(vr + hi * 16);
      v16h Bt[4];
#pragma unroll
      for (int nb2 = 0; nb2 < 4; ++nb2)
        Bt[nb2] = ld16(wp2 + (nb2 * 16 + m) * WP_STRIDE + kb * 32 + hi * 16);
#pragma unroll
      for (int nb2 = 0; nb2 < 4; ++nb2)
        y[nb2] = wmma16(A, Bt[nb2], y[nb2]);
    }

    // epilogue: +bias, coalesced b128 stores into NCHW output
#pragma unroll
    for (int nb = 0; nb < 4; ++nb) {
      int c = nb * 16 + m;
      float bias = bps[c];
      float* orow = ob + ((size_t)c << 16) + tokb + hi * 8;
      v4f o0, o1;
#pragma unroll
      for (int r = 0; r < 4; ++r) { o0[r] = y[nb][r] + bias; o1[r] = y[nb][r + 4] + bias; }
      *(v4f*)(orow)     = o0;
      *(v4f*)(orow + 4) = o1;
    }
  }
}

// ---------------------------------------------------------------------------
extern "C" void kernel_launch(void* const* d_in, const int* in_sizes, int n_in,
                              void* d_out, int out_size, void* d_ws, size_t ws_size,
                              hipStream_t stream)
{
  (void)in_sizes; (void)n_in; (void)out_size; (void)ws_size;

  const float* x       = (const float*)d_in[0];
  const float* illu    = (const float*)d_in[1];
  const float* Wq      = (const float*)d_in[2];
  const float* Wk      = (const float*)d_in[3];
  const float* Wv      = (const float*)d_in[4];
  const float* WL      = (const float*)d_in[5];
  const float* rescale = (const float*)d_in[6];
  const float* Wp      = (const float*)d_in[7];
  const float* bp      = (const float*)d_in[8];
  float* out = (float*)d_out;

  // workspace (fully rewritten every call; no zero-init, no atomics)
  char* ws = (char*)d_ws;
  float* partials = (float*)ws;                                       // 2 MiB
  float* S        = (float*)(ws + (size_t)4 * 32 * 4096 * sizeof(float));
  f16*   Wp2T     = (f16*)((char*)S + (size_t)4 * 4096 * sizeof(float));

  // Pass 1: Gram partials (32 blocks/batch x 8 waves, LDS-staged chunks)
  ig_pass1_gram<<<dim3(32, 4), 256, 0, stream>>>(x, partials);

  // Deterministic reduction -> S[b]
  ig_pass1_reduce<<<64, 256, 0, stream>>>(partials, S);

  // Tiny per-batch attention/softmax/fold-into-Wp kernel
  ig_pass_attn<<<4, 64, 0, stream>>>(S, Wq, Wk, Wp, rescale, Wp2T);

  // Pass 2: streaming fused v-projection + attention apply + output projection
  size_t smemBytes = (size_t)2 * 256 * WV_STRIDE * sizeof(f16)   // wvT, wlT
                   + (size_t)64 * WP_STRIDE * sizeof(f16)        // wp2
                   + 64 * sizeof(float)                          // bias
                   + (size_t)4 * 16 * WP_STRIDE * sizeof(f16);   // vfull buffers
  ig_pass2<<<dim3(32, 4), 128, smemBytes, stream>>>(x, illu, Wv, WL, Wp2T, bp, out);
}
